// EdgeConvGNN_53652731462320
// MI455X (gfx1250) — compile-verified, matrix operand order checked by reading, not run on hardware
//
#include <hip/hip_runtime.h>
#include <math.h>

typedef __attribute__((ext_vector_type(16))) __bf16 v16bf;
typedef __attribute__((ext_vector_type(8)))  float  v8f;
typedef __attribute__((ext_vector_type(8)))  __bf16 bf16x8;

#define HDIM 128
#define HS_STRIDE 136

// ---------------------------------------------------------------------------
// Tiled MLP engine: 256 threads = 8 waves. Tile = 64 rows x 128 out cols.
// Wave w: rows [16*(w&3), +16), cols [64*(w>>2), +64)  -> 4 WMMA n-blocks.
// A (activations) from LDS via 2x ds_load_b128 per fragment.
// B (weights) pre-transposed Wt[n][k] in global: 2x global_load_b128 per
// fragment, L0/L2 resident. No barriers inside the K loop.
//
// CDNA5 16-bit A-frag layout: lane(half,r): elems[0..7]=K{half*8+0..7},
// elems[8..15]=K{half*8+16..23} -> two contiguous 16B runs. B mirrors with
// n = ch + j*16 + r, same K runs -> contiguous in Wt[n][*].
// ---------------------------------------------------------------------------

__device__ __forceinline__ void zero_acc(v8f acc[4]) {
#pragma unroll
  for (int j = 0; j < 4; ++j)
#pragma unroll
    for (int i = 0; i < 8; ++i) acc[j][i] = 0.0f;
}

__device__ __forceinline__ v16bf frag16(const __bf16* p) {
  bf16x8 lo = *(const bf16x8*)p;
  bf16x8 hi = *(const bf16x8*)(p + 16);
  return __builtin_shufflevector(lo, hi, 0, 1, 2, 3, 4, 5, 6, 7, 8, 9, 10, 11,
                                 12, 13, 14, 15);
}

template <int K>
__device__ __forceinline__ void mlp_layer(const __bf16* Xs, int xs_stride,
                                          const __bf16* __restrict__ Wt,
                                          v8f acc[4]) {
  const int tid  = threadIdx.x;
  const int lane = tid & 31;
  const int wave = tid >> 5;
  const int rs   = (wave & 3) << 4;
  const int ch   = (wave >> 2) << 6;
  const int half = lane >> 4;
  const int r    = lane & 15;

  const __bf16* xrow  = &Xs[(rs + r) * xs_stride + (half << 3)];
  const __bf16* wrow0 = &Wt[(size_t)(ch + r) * K + (half << 3)];
  __builtin_prefetch(wrow0, 0, 3);  // global_prefetch_b8: warm weight rows

#pragma unroll
  for (int kb = 0; kb < K / 32; ++kb) {
    v16bf a = frag16(xrow + kb * 32);
#pragma unroll
    for (int j = 0; j < 4; ++j) {
      v16bf b = frag16(wrow0 + (size_t)(j * 16) * K + kb * 32);
      acc[j] = __builtin_amdgcn_wmma_f32_16x16x32_bf16(
          false, a, false, b, (short)0, acc[j], false, false);
    }
  }
}

// hidden = relu(acc + bias) -> bf16 LDS
__device__ __forceinline__ void store_hidden_relu(const v8f acc[4],
                                                  const float* __restrict__ bias,
                                                  __bf16* Hs) {
  const int tid = threadIdx.x, lane = tid & 31, wave = tid >> 5;
  const int rs = (wave & 3) << 4, ch = (wave >> 2) << 6;
  const int half = lane >> 4, r = lane & 15;
#pragma unroll
  for (int j = 0; j < 4; ++j) {
    const int col = ch + (j << 4) + r;
    const float bv = bias[col];
#pragma unroll
    for (int v = 0; v < 8; ++v) {
      const int row = rs + (half << 3) + v;
      float h = acc[j][v] + bv;
      Hs[row * HS_STRIDE + col] = (__bf16)(h > 0.0f ? h : 0.0f);
    }
  }
}

// ---------------------------------------------------------------------------
// Kernels
// ---------------------------------------------------------------------------

// transpose-convert: out[n*K + k] = (bf16) in[k*128 + n]   (N fixed = 128)
__global__ __launch_bounds__(256) void cvt_t_kernel(
    const float* __restrict__ in, __bf16* __restrict__ out, int K) {
  int o = blockIdx.x * 256 + threadIdx.x;
  if (o < K * HDIM) {
    int n = o / K, k = o - n * K;
    out[o] = (__bf16)in[k * HDIM + n];
  }
}

__global__ __launch_bounds__(256) void deg_kernel(
    const int* __restrict__ adj_dst, float* __restrict__ deg, int NA_) {
  int i = blockIdx.x * 256 + threadIdx.x;
  if (i < NA_) atomicAdd(&deg[adj_dst[i]], 1.0f);
}

// encoder: e = (relu(X@W1+b1))@W2+b2, X is [NE,64] f32
__global__ __launch_bounds__(256) void encoder_kernel(
    const float* __restrict__ X, const __bf16* __restrict__ W1t,
    const float* __restrict__ b1, const __bf16* __restrict__ W2t,
    const float* __restrict__ b2, __bf16* __restrict__ e_out, int NE_) {
  __shared__ __bf16 Xs[64 * 72];
  __shared__ __bf16 Hs[64 * HS_STRIDE];
  const int row0 = blockIdx.x * 64, tid = threadIdx.x;
#pragma unroll
  for (int t = 0; t < 16; ++t) {
    int c = tid + t * 256;  // 4096 elems
    int rr = c >> 6, cc = c & 63;
    int gr = row0 + rr;
    float v = (gr < NE_) ? X[(size_t)gr * 64 + cc] : 0.0f;
    Xs[rr * 72 + cc] = (__bf16)v;
  }
  __syncthreads();
  v8f acc[4];
  zero_acc(acc);
  mlp_layer<64>(Xs, 72, W1t, acc);
  store_hidden_relu(acc, b1, Hs);
  __syncthreads();
  v8f acc2[4];
  zero_acc(acc2);
  mlp_layer<128>(Hs, HS_STRIDE, W2t, acc2);

  const int lane = tid & 31, wave = tid >> 5;
  const int rs = (wave & 3) << 4, ch = (wave >> 2) << 6;
  const int half = lane >> 4, r = lane & 15;
#pragma unroll
  for (int j = 0; j < 4; ++j) {
    const int col = ch + (j << 4) + r;
    const float bv = b2[col];
#pragma unroll
    for (int v = 0; v < 8; ++v) {
      const int gr = row0 + rs + (half << 3) + v;
      if (gr < NE_) e_out[(size_t)gr * HDIM + col] = (__bf16)(acc2[j][v] + bv);
    }
  }
}

// msgs for 64 pairs: MLP2(concat(e[dst],e[src])) -> atomicAdd into agg[dst]
__global__ __launch_bounds__(256) void msg_kernel(
    const __bf16* __restrict__ e, const int* __restrict__ adj_dst,
    const int* __restrict__ adj_src, const __bf16* __restrict__ W1t,
    const float* __restrict__ b1, const __bf16* __restrict__ W2t,
    const float* __restrict__ b2, float* __restrict__ agg, int NA_) {
  __shared__ __bf16 Xs[64 * 264];
  __shared__ __bf16 Hs[64 * HS_STRIDE];
  __shared__ int dsts[64];
  const int row0 = blockIdx.x * 64, tid = threadIdx.x;
  if (tid < 64) {
    int p = row0 + tid;
    dsts[tid] = (p < NA_) ? adj_dst[p] : -1;
  }
#pragma unroll
  for (int t = 0; t < 8; ++t) {
    int c = tid + t * 256;  // 2048 chunks of 8 bf16
    int rr = c >> 5, cc = (c & 31) << 3;
    int p = row0 + rr;
    int sr = 0;
    if (p < NA_) sr = (cc < HDIM) ? adj_dst[p] : adj_src[p];
    *(bf16x8*)&Xs[rr * 264 + cc] =
        *(const bf16x8*)&e[(size_t)sr * HDIM + (cc & (HDIM - 1))];
  }
  __syncthreads();
  v8f acc[4];
  zero_acc(acc);
  mlp_layer<256>(Xs, 264, W1t, acc);
  store_hidden_relu(acc, b1, Hs);
  __syncthreads();
  v8f acc2[4];
  zero_acc(acc2);
  mlp_layer<128>(Hs, HS_STRIDE, W2t, acc2);

  const int lane = tid & 31, wave = tid >> 5;
  const int rs = (wave & 3) << 4, ch = (wave >> 2) << 6;
  const int half = lane >> 4, r = lane & 15;
#pragma unroll
  for (int j = 0; j < 4; ++j) {
    const int col = ch + (j << 4) + r;
    const float bv = b2[col];
#pragma unroll
    for (int v = 0; v < 8; ++v) {
      const int row = rs + (half << 3) + v;
      const int dst = dsts[row];
      if (dst >= 0) atomicAdd(&agg[(size_t)dst * HDIM + col], acc2[j][v] + bv);
    }
  }
}

// message = isolated ? MLP2(concat(e,e)) : agg/deg   -> bf16
__global__ __launch_bounds__(256) void selfmsg_kernel(
    const __bf16* __restrict__ e, const float* __restrict__ deg,
    const float* __restrict__ agg, const __bf16* __restrict__ W1t,
    const float* __restrict__ b1, const __bf16* __restrict__ W2t,
    const float* __restrict__ b2, __bf16* __restrict__ msg_out, int NE_) {
  __shared__ __bf16 Xs[64 * 264];
  __shared__ __bf16 Hs[64 * HS_STRIDE];
  const int row0 = blockIdx.x * 64, tid = threadIdx.x;
#pragma unroll
  for (int t = 0; t < 8; ++t) {
    int c = tid + t * 256;
    int rr = c >> 5, cc = (c & 31) << 3;
    int gr = row0 + rr;
    int sr = (gr < NE_) ? gr : 0;
    *(bf16x8*)&Xs[rr * 264 + cc] =
        *(const bf16x8*)&e[(size_t)sr * HDIM + (cc & (HDIM - 1))];
  }
  __syncthreads();
  v8f acc[4];
  zero_acc(acc);
  mlp_layer<256>(Xs, 264, W1t, acc);
  store_hidden_relu(acc, b1, Hs);
  __syncthreads();
  v8f acc2[4];
  zero_acc(acc2);
  mlp_layer<128>(Hs, HS_STRIDE, W2t, acc2);

  const int lane = tid & 31, wave = tid >> 5;
  const int rs = (wave & 3) << 4, ch = (wave >> 2) << 6;
  const int half = lane >> 4, r = lane & 15;
#pragma unroll
  for (int j = 0; j < 4; ++j) {
    const int col = ch + (j << 4) + r;
    const float bv = b2[col];
#pragma unroll
    for (int v = 0; v < 8; ++v) {
      const int gr = row0 + rs + (half << 3) + v;
      if (gr < NE_) {
        float d = deg[gr];
        float val = (d == 0.0f) ? (acc2[j][v] + bv)
                                : agg[(size_t)gr * HDIM + col] / d;
        msg_out[(size_t)gr * HDIM + col] = (__bf16)val;
      }
    }
  }
}

// e_new = relu(MLP2(concat(e, message)))
__global__ __launch_bounds__(256) void update_kernel(
    const __bf16* __restrict__ e, const __bf16* __restrict__ msg,
    const __bf16* __restrict__ W1t, const float* __restrict__ b1,
    const __bf16* __restrict__ W2t, const float* __restrict__ b2,
    __bf16* __restrict__ e_out, int NE_) {
  __shared__ __bf16 Xs[64 * 264];
  __shared__ __bf16 Hs[64 * HS_STRIDE];
  const int row0 = blockIdx.x * 64, tid = threadIdx.x;
#pragma unroll
  for (int t = 0; t < 8; ++t) {
    int c = tid + t * 256;
    int rr = c >> 5, cc = (c & 31) << 3;
    int gr = row0 + rr;
    int sr = (gr < NE_) ? gr : 0;
    const __bf16* base = (cc < HDIM) ? e : msg;
    *(bf16x8*)&Xs[rr * 264 + cc] =
        *(const bf16x8*)&base[(size_t)sr * HDIM + (cc & (HDIM - 1))];
  }
  __syncthreads();
  v8f acc[4];
  zero_acc(acc);
  mlp_layer<256>(Xs, 264, W1t, acc);
  store_hidden_relu(acc, b1, Hs);
  __syncthreads();
  v8f acc2[4];
  zero_acc(acc2);
  mlp_layer<128>(Hs, HS_STRIDE, W2t, acc2);

  const int lane = tid & 31, wave = tid >> 5;
  const int rs = (wave & 3) << 4, ch = (wave >> 2) << 6;
  const int half = lane >> 4, r = lane & 15;
#pragma unroll
  for (int j = 0; j < 4; ++j) {
    const int col = ch + (j << 4) + r;
    const float bv = b2[col];
#pragma unroll
    for (int v = 0; v < 8; ++v) {
      const int gr = row0 + rs + (half << 3) + v;
      if (gr < NE_) {
        float h = acc2[j][v] + bv;
        e_out[(size_t)gr * HDIM + col] = (__bf16)(h > 0.0f ? h : 0.0f);
      }
    }
  }
}

// predictor: 2 WMMA layers + f32 matvec + sigmoid
__global__ __launch_bounds__(256) void pred_kernel(
    const __bf16* __restrict__ e, const int* __restrict__ ci,
    const int* __restrict__ cj, const __bf16* __restrict__ W1t,
    const float* __restrict__ b1, const __bf16* __restrict__ W2t,
    const float* __restrict__ b2, const float* __restrict__ W3,
    const float* __restrict__ b3, float* __restrict__ probs,
    float* __restrict__ logits, int NC_) {
  __shared__ __bf16 Xs[64 * 264];
  __shared__ __bf16 Hs[64 * HS_STRIDE];
  const int row0 = blockIdx.x * 64, tid = threadIdx.x;
#pragma unroll
  for (int t = 0; t < 8; ++t) {
    int c = tid + t * 256;
    int rr = c >> 5, cc = (c & 31) << 3;
    int p = row0 + rr;
    int sr = 0;
    if (p < NC_) sr = (cc < HDIM) ? ci[p] : cj[p];
    *(bf16x8*)&Xs[rr * 264 + cc] =
        *(const bf16x8*)&e[(size_t)sr * HDIM + (cc & (HDIM - 1))];
  }
  __syncthreads();
  v8f acc[4];
  zero_acc(acc);
  mlp_layer<256>(Xs, 264, W1t, acc);
  store_hidden_relu(acc, b1, Hs);
  __syncthreads();
  v8f acc2[4];
  zero_acc(acc2);
  mlp_layer<128>(Hs, HS_STRIDE, W2t, acc2);
  __syncthreads();  // all Xs reads (layer1) done on all waves

  // h2 = relu(acc2 + b2) stored f32, reusing Xs space (64 x 132 floats)
  float* Hf = (float*)Xs;
  const int lane = tid & 31, wave = tid >> 5;
  const int rs = (wave & 3) << 4, ch = (wave >> 2) << 6;
  const int half = lane >> 4, r = lane & 15;
#pragma unroll
  for (int j = 0; j < 4; ++j) {
    const int col = ch + (j << 4) + r;
    const float bv = b2[col];
#pragma unroll
    for (int v = 0; v < 8; ++v) {
      const int row = rs + (half << 3) + v;
      float h = acc2[j][v] + bv;
      Hf[row * 132 + col] = h > 0.0f ? h : 0.0f;
    }
  }
  __syncthreads();
  if (tid < 64) {
    int gr = row0 + tid;
    if (gr < NC_) {
      float s = b3[0];
#pragma unroll 4
      for (int k = 0; k < HDIM; ++k) s += Hf[tid * 132 + k] * W3[k];
      logits[gr] = s;
      probs[gr] = 1.0f / (1.0f + expf(-s));
    }
  }
}

// ---------------------------------------------------------------------------
// Host launch
// ---------------------------------------------------------------------------

extern "C" void kernel_launch(void* const* d_in, const int* in_sizes, int n_in,
                              void* d_out, int out_size, void* d_ws,
                              size_t ws_size, hipStream_t stream) {
  const float* edge_features = (const float*)d_in[0];
  const int* adj_dst = (const int*)d_in[1];
  const int* adj_src = (const int*)d_in[2];
  const int* cand_i = (const int*)d_in[3];
  const int* cand_j = (const int*)d_in[4];
  const float* enc_W1 = (const float*)d_in[5];
  const float* enc_b1 = (const float*)d_in[6];
  const float* enc_W2 = (const float*)d_in[7];
  const float* enc_b2 = (const float*)d_in[8];
  const float* msg_W1 = (const float*)d_in[9];
  const float* msg_b1 = (const float*)d_in[10];
  const float* msg_W2 = (const float*)d_in[11];
  const float* msg_b2 = (const float*)d_in[12];
  const float* upd_W1 = (const float*)d_in[13];
  const float* upd_b1 = (const float*)d_in[14];
  const float* upd_W2 = (const float*)d_in[15];
  const float* upd_b2 = (const float*)d_in[16];
  const float* pred_W1 = (const float*)d_in[17];
  const float* pred_b1 = (const float*)d_in[18];
  const float* pred_W2 = (const float*)d_in[19];
  const float* pred_b2 = (const float*)d_in[20];
  const float* pred_W3 = (const float*)d_in[21];
  const float* pred_b3 = (const float*)d_in[22];

  const int NE = in_sizes[0] / 64;
  const int NA = in_sizes[1];
  const int NC = in_sizes[3];
  const int L = 3;

  char* ws = (char*)d_ws;
  size_t off = 0;
  auto carve = [&](size_t bytes) -> void* {
    void* p = ws + off;
    off = (off + bytes + 255) & ~(size_t)255;
    return p;
  };
  __bf16* e0 = (__bf16*)carve((size_t)NE * HDIM * 2);
  __bf16* e1 = (__bf16*)carve((size_t)NE * HDIM * 2);
  __bf16* msgb = (__bf16*)carve((size_t)NE * HDIM * 2);
  float* agg = (float*)carve((size_t)NE * HDIM * 4);
  float* deg = (float*)carve((size_t)NE * 4);
  __bf16* encW1t = (__bf16*)carve((size_t)in_sizes[5] * 2);
  __bf16* encW2t = (__bf16*)carve((size_t)in_sizes[7] * 2);
  __bf16* msgW1t = (__bf16*)carve((size_t)in_sizes[9] * 2);
  __bf16* msgW2t = (__bf16*)carve((size_t)in_sizes[11] * 2);
  __bf16* updW1t = (__bf16*)carve((size_t)in_sizes[13] * 2);
  __bf16* updW2t = (__bf16*)carve((size_t)in_sizes[15] * 2);
  __bf16* predW1t = (__bf16*)carve((size_t)in_sizes[17] * 2);
  __bf16* predW2t = (__bf16*)carve((size_t)in_sizes[19] * 2);

  // transpose-convert weights: Wt[n*K + k] = W[k*128 + n]
  auto cvt_t = [&](const float* src, __bf16* dst, int K) {
    int n = K * HDIM;
    cvt_t_kernel<<<(n + 255) / 256, 256, 0, stream>>>(src, dst, K);
  };
  cvt_t(enc_W1, encW1t, 64);
  cvt_t(enc_W2, encW2t, HDIM);
  for (int l = 0; l < L; ++l) {
    cvt_t(msg_W1 + (size_t)l * 2 * HDIM * HDIM, msgW1t + (size_t)l * 2 * HDIM * HDIM, 2 * HDIM);
    cvt_t(msg_W2 + (size_t)l * HDIM * HDIM, msgW2t + (size_t)l * HDIM * HDIM, HDIM);
    cvt_t(upd_W1 + (size_t)l * 2 * HDIM * HDIM, updW1t + (size_t)l * 2 * HDIM * HDIM, 2 * HDIM);
    cvt_t(upd_W2 + (size_t)l * HDIM * HDIM, updW2t + (size_t)l * HDIM * HDIM, HDIM);
  }
  cvt_t(pred_W1, predW1t, 2 * HDIM);
  cvt_t(pred_W2, predW2t, HDIM);

  hipMemsetAsync(deg, 0, (size_t)NE * 4, stream);
  deg_kernel<<<(NA + 255) / 256, 256, 0, stream>>>(adj_dst, deg, NA);

  const int gNE = (NE + 63) / 64;
  const int gNA = (NA + 63) / 64;
  const int gNC = (NC + 63) / 64;

  encoder_kernel<<<gNE, 256, 0, stream>>>(edge_features, encW1t, enc_b1,
                                          encW2t, enc_b2, e0, NE);

  __bf16* ecur = e0;
  __bf16* enext = e1;
  for (int l = 0; l < L; ++l) {
    const __bf16* mW1 = msgW1t + (size_t)l * 2 * HDIM * HDIM;
    const __bf16* mW2 = msgW2t + (size_t)l * HDIM * HDIM;
    const float* mb1 = msg_b1 + (size_t)l * HDIM;
    const float* mb2 = msg_b2 + (size_t)l * HDIM;
    const __bf16* uW1 = updW1t + (size_t)l * 2 * HDIM * HDIM;
    const __bf16* uW2 = updW2t + (size_t)l * HDIM * HDIM;
    const float* ub1 = upd_b1 + (size_t)l * HDIM;
    const float* ub2 = upd_b2 + (size_t)l * HDIM;

    hipMemsetAsync(agg, 0, (size_t)NE * HDIM * 4, stream);
    msg_kernel<<<gNA, 256, 0, stream>>>(ecur, adj_dst, adj_src, mW1, mb1, mW2,
                                        mb2, agg, NA);
    selfmsg_kernel<<<gNE, 256, 0, stream>>>(ecur, deg, agg, mW1, mb1, mW2, mb2,
                                            msgb, NE);
    update_kernel<<<gNE, 256, 0, stream>>>(ecur, msgb, uW1, ub1, uW2, ub2,
                                           enext, NE);
    __bf16* t = ecur;
    ecur = enext;
    enext = t;
  }

  float* probs = (float*)d_out;
  float* logits = probs + NC;
  pred_kernel<<<gNC, 256, 0, stream>>>(ecur, cand_i, cand_j, predW1t, pred_b1,
                                       predW2t, pred_b2, pred_W3, pred_b3,
                                       probs, logits, NC);
}
